// EncoderLayer_28595892256994
// MI455X (gfx1250) — matrix operand order repaired and checked by reading, not compiled, not run on hardware
//
#include <hip/hip_runtime.h>

typedef _Float16 v16h __attribute__((ext_vector_type(16)));
typedef _Float16 v8h  __attribute__((ext_vector_type(8)));
typedef _Float16 v4h  __attribute__((ext_vector_type(4)));
typedef float    v8f  __attribute__((ext_vector_type(8)));

#define NDIR 6
#define IDIM 256
#define KDIM 512
#define ODIM 512
#define BB   8
#define LL   4096
#define NCHN 8192
#define NTOK (BB*LL)     // 32768 tokens
#define MTILE 32

// ---------------- workspace layout ----------------
// [0, 3MB)          : W_h, f16 weights (NDIR*ODIM*KDIM halves = 3 MB)
// [3MB, 3MB+32)     : counts (6 ints)
// [3MB+256, +768KB) : buckets (NDIR * NTOK ints)
#define WS_WH_BYTES   (3u*1024u*1024u)
#define WS_CNT_OFF    WS_WH_BYTES
#define WS_BKT_OFF    (WS_WH_BYTES + 256u)

__global__ void k_zero(int* __restrict__ counts) {
    if (threadIdx.x < NDIR) counts[threadIdx.x] = 0;
}

// fp32 -> f16 weight conversion: 6*512*512 = 1,572,864 elems, 4 per thread
__global__ void k_cvt_w(const float* __restrict__ W, _Float16* __restrict__ Wh) {
    const int i = (blockIdx.x * 256 + threadIdx.x) * 4;
    float4 f = *(const float4*)(W + i);
    v4h h;
    h.x = (_Float16)f.x; h.y = (_Float16)f.y;
    h.z = (_Float16)f.z; h.w = (_Float16)f.w;
    *(v4h*)(Wh + i) = h;
}

// bucket token ids by direction
__global__ void k_bin(const int* __restrict__ vec, const int* __restrict__ dmap,
                      int* __restrict__ counts, int* __restrict__ buckets) {
    const int t = blockIdx.x * 256 + threadIdx.x;   // t in [0, NTOK)
    const int dir = dmap[vec[t]];
    const int pos = atomicAdd(&counts[dir], 1);
    buckets[dir * NTOK + pos] = t;
}

// grouped GEMM: one block = 32 tokens (same dir) x full ODIM(512), K = 512
__global__ __launch_bounds__(256) void k_gemm(
    const float*    __restrict__ last,
    const _Float16* __restrict__ Wh,
    const float*    __restrict__ bmerge,
    const float*    __restrict__ amerge,
    const int*      __restrict__ child_l,
    const int*      __restrict__ child_r,
    const int*      __restrict__ vec,
    const int*      __restrict__ drev,
    const int*      __restrict__ counts,
    const int*      __restrict__ buckets,
    float*          __restrict__ out)
{
    __shared__ _Float16 sA[MTILE][KDIM + 8];   // +8 halves pad: conflict-free frag reads
    __shared__ int s_tok[MTILE];

    const int dir   = blockIdx.y;
    const int cnt   = counts[dir];
    const int mbase = blockIdx.x * MTILE;
    if (mbase >= cnt) return;
    const int rows = min(MTILE, cnt - mbase);

    const int tid = threadIdx.x;

    // ---- stage A tile: gather + swap + fp32->f16 into LDS ----
    {
        const int r = tid >> 3;          // 0..31 : token row
        const int c = tid & 7;           // 0..7  : 64-elem chunk
        _Float16* dst = &sA[r][c * 64];
        if (r < rows) {
            const int t = buckets[dir * NTOK + mbase + r];
            if (c == 0) s_tok[r] = t;
            const int b  = t >> 12;          // L = 4096
            const int l  = t & (LL - 1);
            const int d  = drev[vec[t]];
            const int cl = child_l[l];
            const int cr = child_r[l];
            const long rowFirst  = ((long)b * NCHN + (d ? cr : cl)) * IDIM;
            const long rowSecond = ((long)b * NCHN + (d ? cl : cr)) * IDIM;
            const float* src = last + (c < 4 ? rowFirst : rowSecond) + (long)(c & 3) * 64;
            #pragma unroll
            for (int i = 0; i < 64; i += 4) {
                float4 f = *(const float4*)(src + i);
                v4h h;
                h.x = (_Float16)f.x; h.y = (_Float16)f.y;
                h.z = (_Float16)f.z; h.w = (_Float16)f.w;
                *(v4h*)(dst + i) = h;
            }
        } else {
            if (c == 0) s_tok[r] = 0;
            v4h z = {};
            #pragma unroll
            for (int i = 0; i < 64; i += 4) *(v4h*)(dst + i) = z;
        }
    }
    __syncthreads();

    const int wave = tid >> 5;       // 8 waves (wave32)
    const int lane = tid & 31;
    const int wm   = wave & 1;       // m-tile 0/1 (rows 16*wm .. +15)
    const int wn   = wave >> 1;      // 0..3 -> columns [wn*128, wn*128+128)
    const int lmod = lane & 15;
    const int hi   = lane >> 4;      // 0: lanes 0-15, 1: lanes 16-31

    v8f acc[8] = {};

    const _Float16* arow  = &sA[wm * 16 + lmod][0];
    const _Float16* wbase = Wh + (size_t)dir * ODIM * KDIM;

    for (int kk = 0; kk < KDIM; kk += 32) {
        // A fragment, contiguous K (matches B):
        // lane<16 : K = kk..kk+15
        // lane>=16: K = kk+16..kk+31
        const int a0off = kk + hi * 16;
        v8h a0 = *(const v8h*)(arow + a0off);
        v8h a1 = *(const v8h*)(arow + a0off + 8);
        v16h afrag;
        #pragma unroll
        for (int i = 0; i < 8; i++) { afrag[i] = a0[i]; afrag[8 + i] = a1[i]; }

        // B fragment: lane holds column N = 16*nt + lmod, 16 consecutive K
        const int koff = kk + hi * 16;
        #pragma unroll
        for (int nt = 0; nt < 8; nt++) {
            const int n = wn * 128 + nt * 16 + lmod;
            v16h bfrag = *(const v16h*)(wbase + (size_t)n * KDIM + koff);
            acc[nt] = __builtin_amdgcn_wmma_f32_16x16x32_f16(
                false, afrag, false, bfrag, (short)0, acc[nt], false, false);
        }
    }

    // ---- epilogue: bias + LeakyReLU + scatter rows to tokens ----
    const float al = amerge[dir];
    #pragma unroll
    for (int nt = 0; nt < 8; nt++) {
        const int n = wn * 128 + nt * 16 + lmod;
        const float bias = bmerge[dir * ODIM + n];
        #pragma unroll
        for (int r = 0; r < 8; r++) {
            const int m = wm * 16 + r + hi * 8;   // C layout: VGPR r = row r (+8 for hi lanes)
            if (m < rows) {
                float y = acc[nt][r] + bias;
                y = (y > 0.0f) ? y : al * y;
                out[(size_t)s_tok[m] * ODIM + n] = y;
            }
        }
    }
}

extern "C" void kernel_launch(void* const* d_in, const int* in_sizes, int n_in,
                              void* d_out, int out_size, void* d_ws, size_t ws_size,
                              hipStream_t stream) {
    const float* last  = (const float*)d_in[0];
    const float* W     = (const float*)d_in[1];
    const float* bmrg  = (const float*)d_in[2];
    const float* amrg  = (const float*)d_in[3];
    const int*   cl    = (const int*)d_in[4];
    const int*   cr    = (const int*)d_in[5];
    const int*   vec   = (const int*)d_in[6];
    const int*   drev  = (const int*)d_in[7];
    const int*   dmap  = (const int*)d_in[8];
    float*       out   = (float*)d_out;

    char* ws = (char*)d_ws;
    _Float16* Wh   = (_Float16*)ws;
    int* counts    = (int*)(ws + WS_CNT_OFF);
    int* buckets   = (int*)(ws + WS_BKT_OFF);

    k_zero<<<1, 32, 0, stream>>>(counts);
    k_cvt_w<<<(NDIR * ODIM * KDIM) / (256 * 4), 256, 0, stream>>>(W, Wh);
    k_bin<<<NTOK / 256, 256, 0, stream>>>(vec, dmap, counts, buckets);

    dim3 grid(NTOK / MTILE, NDIR);   // 1024 x 6, early-exit past bucket count
    k_gemm<<<grid, 256, 0, stream>>>(last, Wh, bmrg, amrg, cl, cr, vec, drev,
                                     counts, buckets, out);
}
